// EdgeConv_87162066305548
// MI455X (gfx1250) — matrix operand order repaired and checked by reading, not compiled, not run on hardware
//
#include <hip/hip_runtime.h>

typedef _Float16 half8 __attribute__((ext_vector_type(8)));
typedef _Float16 v16h  __attribute__((ext_vector_type(16)));
typedef float    v8f   __attribute__((ext_vector_type(8)));

#define BATCH 16
#define NPTS  1024
#define KNN_K 16
#define FEAT  64
#define HID   64
#define LN_EPS 1e-5f

// ---------------------------------------------------------------------------
// KNN: one thread per query point; register insertion-sort of top-16 squared
// distances (sqrt is monotone, ordering identical; ties resolve to the lower
// index first, matching jax.lax.top_k's stable tie-break).
// ---------------------------------------------------------------------------
__global__ void knn_kernel(const float* __restrict__ pts,
                           const int*   __restrict__ mask,
                           int*         __restrict__ knn) {
    int t = blockIdx.x * blockDim.x + threadIdx.x;
    if (t >= BATCH * NPTS) return;
    int b = t >> 10, n = t & (NPTS - 1);
    const float* pb = pts + (size_t)b * NPTS * 3;
    const int*   mb = mask + (size_t)b * NPTS;
    float qx = pb[n * 3 + 0], qy = pb[n * 3 + 1], qz = pb[n * 3 + 2];

    float bd[KNN_K];
    int   bi[KNN_K];
#pragma unroll
    for (int i = 0; i < KNN_K; ++i) { bd[i] = __builtin_inff(); bi[i] = 0; }

    for (int j = 0; j < NPTS; ++j) {
        if (!mb[j]) continue;                       // masked -> dist = inf
        float dx = qx - pb[j * 3 + 0];
        float dy = qy - pb[j * 3 + 1];
        float dz = qz - pb[j * 3 + 2];
        float d = dx * dx + dy * dy + dz * dz;
        if (d < bd[KNN_K - 1]) {                    // strict: keep earlier idx on tie
            bd[KNN_K - 1] = d; bi[KNN_K - 1] = j;
#pragma unroll
            for (int i = KNN_K - 1; i > 0; --i) {
                if (bd[i] < bd[i - 1]) {            // strict: stable for equal dists
                    float td = bd[i]; bd[i] = bd[i - 1]; bd[i - 1] = td;
                    int   ti = bi[i]; bi[i] = bi[i - 1]; bi[i - 1] = ti;
                }
            }
        }
    }
    int* o = knn + (size_t)t * KNN_K;
#pragma unroll
    for (int i = 0; i < KNN_K; ++i) o[i] = bi[i];
}

// ---------------------------------------------------------------------------
// Convert weights to f16 once into workspace: W0 (64x128) then Wh (4x64x64),
// both stored [out][in] so B-fragments read two runs of 8 contiguous halves along K.
// ---------------------------------------------------------------------------
__global__ void convw_kernel(const float* __restrict__ W0,
                             const float* __restrict__ Wh,
                             _Float16*    __restrict__ Wg) {
    int t = blockIdx.x * blockDim.x + threadIdx.x;
    if (t < HID * 2 * FEAT) {
        Wg[t] = (_Float16)W0[t];
    } else if (t < HID * 2 * FEAT + 4 * HID * HID) {
        Wg[t] = (_Float16)Wh[t - HID * 2 * FEAT];
    }
}

// ---------------------------------------------------------------------------
// n_tracks[b] = sum(mask[b,:])
// ---------------------------------------------------------------------------
__global__ void ntracks_kernel(const int* __restrict__ mask,
                               float*     __restrict__ nt) {
    __shared__ int sdata[256];
    int b = blockIdx.x, t = threadIdx.x;
    int s = 0;
    for (int j = t; j < NPTS; j += 256) s += mask[(size_t)b * NPTS + j];
    sdata[t] = s;
    __syncthreads();
    for (int off = 128; off > 0; off >>= 1) {
        if (t < off) sdata[t] += sdata[t + off];
        __syncthreads();
    }
    if (t == 0) nt[b] = (float)sdata[0];
}

// ---------------------------------------------------------------------------
// One WMMA layer: D[16 x 64] = A[16 x KIN] * W^T + bias.
// A-fragment per ISA 16-bit 16x32 layout: lanes 0-15 -> K {kb*32+0..7, +16..23},
// lanes 16-31 -> K {+8..15, +24..31}. B-fragment per 32x16 layout: lane = col,
// lane>=16 holds K 8..15, 24..31 (same order as A). D layout: VGPR r holds (M = r + 8*(lane>>4), N = lane&15).
// ---------------------------------------------------------------------------
template <int KIN>
__device__ __forceinline__ void layer_gemm(const _Float16* __restrict__ W,
                                           const float*    __restrict__ bias,
                                           const _Float16* Ah, float* Hf, int lane) {
    constexpr int NKB = KIN / 32;
    const int row  = lane & 15;
    const int khi  = (lane >> 4) * 8;    // A: hi-lane K offset
    const int koff = (lane >> 4) * 8;    // B: hi-lane K offset
    const int mrow = (lane >> 4) * 8;    // D: hi-lane M offset

    v16h afrag[NKB];
#pragma unroll
    for (int kb = 0; kb < NKB; ++kb) {
        const half8* p = (const half8*)(Ah + row * KIN + kb * 32 + khi);
        half8 lo = p[0];
        half8 hi = p[2];                 // +16 halves
#pragma unroll
        for (int i = 0; i < 8; ++i) { afrag[kb][i] = lo[i]; afrag[kb][8 + i] = hi[i]; }
    }

#pragma unroll
    for (int ct = 0; ct < 4; ++ct) {
        const int col = ct * 16 + (lane & 15);
        v8f acc = {0.f, 0.f, 0.f, 0.f, 0.f, 0.f, 0.f, 0.f};
#pragma unroll
        for (int kb = 0; kb < NKB; ++kb) {
            const half8* q = (const half8*)(W + col * KIN + kb * 32 + koff);
            half8 b0 = q[0], b1 = q[2];
            v16h bfrag;
#pragma unroll
            for (int i = 0; i < 8; ++i) { bfrag[i] = b0[i]; bfrag[8 + i] = b1[i]; }
            acc = __builtin_amdgcn_wmma_f32_16x16x32_f16(
                false, afrag[kb], false, bfrag, (short)0, acc, false, false);
        }
        const float bv = bias[col];
#pragma unroll
        for (int r = 0; r < 8; ++r)
            Hf[(mrow + r) * HID + col] = acc[r] + bv;
    }
}

// LayerNorm + PReLU on a 16x64 fp32 tile; lanes 0-15 own one row each.
__device__ __forceinline__ void ln_prelu(float* Hf, _Float16* Ah,
                                         const float* __restrict__ g,
                                         const float* __restrict__ beta,
                                         float a, int lane, bool toHalf) {
    if (lane < 16) {
        float* rowp = Hf + lane * HID;
        float s = 0.f, s2 = 0.f;
        for (int c = 0; c < HID; ++c) { float v = rowp[c]; s += v; s2 += v * v; }
        const float mu   = s * (1.f / HID);
        const float var  = s2 * (1.f / HID) - mu * mu;
        const float rstd = rsqrtf(var + LN_EPS);
        for (int c = 0; c < HID; ++c) {
            float v = g[c] * ((rowp[c] - mu) * rstd) + beta[c];
            v = v > 0.f ? v : a * v;
            if (toHalf) Ah[lane * HID + c] = (_Float16)v;
            else        rowp[c] = v;
        }
    }
}

// ---------------------------------------------------------------------------
// MLP: one wave32 block per (b, n). The K=16 edges form the M dim of the
// WMMA tiles. All intermediates stay in LDS; 48 v_wmma per point.
// ---------------------------------------------------------------------------
__global__ void __launch_bounds__(32)
mlp_kernel(const float*    __restrict__ x,
           const int*      __restrict__ knn,
           const _Float16* __restrict__ Wg,
           const float*    __restrict__ b0,
           const float*    __restrict__ bh,
           const float*    __restrict__ ln_g,
           const float*    __restrict__ ln_b,
           const float*    __restrict__ pa,
           const float*    __restrict__ nt,
           float*          __restrict__ out) {
    __shared__ _Float16 Ah[16 * 128];   // f16 activations (layer input)
    __shared__ float    Hf[16 * HID];   // f32 GEMM output / LN staging

    const int wg = blockIdx.x;
    const int b  = wg >> 10;
    const int n  = wg & (NPTS - 1);
    const int lane = threadIdx.x;

    // Build layer-0 input: h[k] = [ x(b,n) , x(b,n) - x(b, idx[k]) ]  (16 x 128)
    {
        const int row  = lane >> 1;
        const int hsel = lane & 1;
        const float* xq = x + ((size_t)b * NPTS + n) * FEAT;
        if (hsel == 0) {
            for (int c = 0; c < FEAT; ++c)
                Ah[row * 128 + c] = (_Float16)xq[c];
        } else {
            const int j = knn[((size_t)b * NPTS + n) * KNN_K + row];
            const float* xn = x + ((size_t)b * NPTS + j) * FEAT;
            for (int c = 0; c < FEAT; ++c)
                Ah[row * 128 + 64 + c] = (_Float16)(xq[c] - xn[c]);
        }
    }
    __syncthreads();

    // Block 0: 128 -> 64
    layer_gemm<128>(Wg, b0, Ah, Hf, lane);
    __syncthreads();
    ln_prelu(Hf, Ah, ln_g, ln_b, pa[0], lane, /*toHalf=*/true);
    __syncthreads();

    // Blocks 1..4: 64 -> 64
#pragma unroll
    for (int i = 0; i < 4; ++i) {
        const _Float16* W = Wg + HID * 2 * FEAT + i * HID * HID;
        layer_gemm<64>(W, bh + i * HID, Ah, Hf, lane);
        __syncthreads();
        ln_prelu(Hf, Ah, ln_g + (i + 1) * HID, ln_b + (i + 1) * HID,
                 pa[i + 1], lane, /*toHalf=*/(i < 3));   // last block stays fp32 in Hf
        __syncthreads();
    }

    // Masked mean over k: nm[k] = (k < n_tracks); divide by n_safe.
    const float ntv  = nt[b];
    const int   kmax = ntv < (float)KNN_K ? (int)ntv : KNN_K;
    const float inv  = 1.0f / (ntv == 0.0f ? 1.0f : ntv);
    for (int c = lane; c < HID; c += 32) {
        float s = 0.f;
        for (int k = 0; k < kmax; ++k) s += Hf[k * HID + c];
        out[((size_t)b * NPTS + n) * HID + c] = s * inv;
    }
}

// ---------------------------------------------------------------------------
extern "C" void kernel_launch(void* const* d_in, const int* in_sizes, int n_in,
                              void* d_out, int out_size, void* d_ws, size_t ws_size,
                              hipStream_t stream) {
    const float* points = (const float*)d_in[0];
    const float* x      = (const float*)d_in[1];
    const int*   mask   = (const int*)  d_in[2];
    const float* W0     = (const float*)d_in[3];
    const float* b0     = (const float*)d_in[4];
    const float* Wh     = (const float*)d_in[5];
    const float* bh     = (const float*)d_in[6];
    const float* ln_g   = (const float*)d_in[7];
    const float* ln_b   = (const float*)d_in[8];
    const float* pa     = (const float*)d_in[9];
    float* out = (float*)d_out;

    char* ws = (char*)d_ws;
    _Float16* Wg  = (_Float16*)ws;                               // 24576 halves (48 KB)
    int*      knn = (int*)(ws + 65536);                          // 262144 ints (1 MB)
    float*    nt  = (float*)(ws + 65536 + (size_t)BATCH * NPTS * KNN_K * 4);

    convw_kernel  <<<96, 256, 0, stream>>>(W0, Wh, Wg);
    ntracks_kernel<<<BATCH, 256, 0, stream>>>(mask, nt);
    knn_kernel    <<<(BATCH * NPTS) / 256, 256, 0, stream>>>(points, mask, knn);
    mlp_kernel    <<<BATCH * NPTS, 32, 0, stream>>>(x, knn, Wg, b0, bh,
                                                    ln_g, ln_b, pa, nt, out);
}
